// MultiHeadAttention_4157528342865
// MI455X (gfx1250) — compile-verified, matrix operand order checked by reading, not run on hardware
//
#include <hip/hip_runtime.h>
#include <hip/hip_bf16.h>

typedef __attribute__((ext_vector_type(16))) _Float16 v16h;
typedef __attribute__((ext_vector_type(8)))  float    v8f;

#define D_MODEL 768
#define N_HEADS 12
#define D_K     64
#define SEQ     2048
#define BATCH   2
#define M_TOT   (BATCH*SEQ)   // 4096

union AFrag { v16h v; uint4 q[2]; };
union CFrag { v8f  v; float f[8]; };

// ---------------------------------------------------------------- f32 -> f16
__global__ void cvt_f32_f16(const float* __restrict__ src,
                            _Float16* __restrict__ dst, int n) {
  int i = blockIdx.x * blockDim.x + threadIdx.x;
  int stride = gridDim.x * blockDim.x;
  for (; i < n; i += stride) dst[i] = (_Float16)src[i];
}

// f32 [K,N] -> f16 [N,K] (weight transpose so GEMM B-fragments are
// contiguous 16-byte loads; one-time 1.2 MB cost per weight matrix)
__global__ void cvt_transpose_f32_f16(const float* __restrict__ src,
                                      _Float16* __restrict__ dst) {
  int i = blockIdx.x * blockDim.x + threadIdx.x;
  int stride = gridDim.x * blockDim.x;
  const int n = D_MODEL * D_MODEL;
  for (; i < n; i += stride) {
    int k = i / D_MODEL, c = i % D_MODEL;          // coalesced read
    dst[(size_t)c * D_MODEL + k] = (_Float16)src[i];
  }
}

// ---------------------------------------------------------------- WMMA GEMM
// C[M,N] = A[M,K] * Wt^T + bias, with Wt stored [N,K] f16 row-major.
// Block: 8 waves, each wave owns a 16(M) x 64(N) strip: one A-fragment is
// reused against 4 B-fragments per k-step (4 WMMAs / 10 b128 loads).
// Block tile: 128(M) x 64(N). All fragments loaded straight from global
// (weights live in the 192 MB L2; no LDS staging needed).
// mode 0: f16 out, [B,H,S,dk] (Q / K)
// mode 1: f16 out, [B,H,dk,S] (V transposed, for contiguous P*V B-fragments)
// mode 2: f32 out, row-major [M, 768] (final projection)
__global__ __launch_bounds__(256)
void gemm_wmma(const _Float16* __restrict__ A,    // [M, 768] f16
               const _Float16* __restrict__ Wt,   // [768(N), 768(K)] f16
               const float* __restrict__ bias,    // [768]
               void* __restrict__ outp, int mode) {
  const int tid  = threadIdx.x;
  const int wave = tid >> 5;
  const int lane = tid & 31;
  const int lr   = lane & 15;
  const int hi   = lane >> 4;

  const int m0 = blockIdx.x * 128 + wave * 16;
  const int n0 = blockIdx.y * 64;

  CFrag acc[4];
#pragma unroll
  for (int nt = 0; nt < 4; ++nt)
#pragma unroll
    for (int i = 0; i < 8; ++i) acc[nt].f[i] = 0.0f;

  // Per-lane row pointers: A row = output row, Wt row = output column.
  const _Float16* arow = A + (size_t)(m0 + lr) * D_MODEL;
  const _Float16* brow0 = Wt + (size_t)(n0 + lr) * D_MODEL;

  for (int k0 = 0; k0 < D_MODEL; k0 += 32) {
    AFrag a;
    a.q[0] = *(const uint4*)(arow + k0 + hi * 8);
    a.q[1] = *(const uint4*)(arow + k0 + 16 + hi * 8);
    if (k0 + 32 < D_MODEL)
      __builtin_prefetch(arow + k0 + 32, 0, 1);
#pragma unroll
    for (int nt = 0; nt < 4; ++nt) {
      const _Float16* brow = brow0 + (size_t)(nt * 16) * D_MODEL;
      AFrag b;
      b.q[0] = *(const uint4*)(brow + k0 + hi * 8);
      b.q[1] = *(const uint4*)(brow + k0 + 16 + hi * 8);
      acc[nt].v = __builtin_amdgcn_wmma_f32_16x16x32_f16(
          false, a.v, false, b.v, (short)0, acc[nt].v, false, false);
    }
  }

#pragma unroll
  for (int nt = 0; nt < 4; ++nt) {
#pragma unroll
    for (int r = 0; r < 8; ++r) {
      int row = m0 + r + 8 * hi;           // C layout: M = r + 8*(lane/16)
      int col = n0 + nt * 16 + lr;         // N = lane%16
      float val = acc[nt].f[r] + bias[col];
      if (mode == 2) {
        ((float*)outp)[(size_t)row * D_MODEL + col] = val;
      } else {
        int b = row >> 11, s = row & (SEQ - 1);
        int h = col >> 6,  d = col & (D_K - 1);
        _Float16* o = (_Float16*)outp;
        if (mode == 0)
          o[((size_t)(b * N_HEADS + h) * SEQ + s) * D_K + d] = (_Float16)val;
        else
          o[((size_t)(b * N_HEADS + h) * D_K + d) * SEQ + s] = (_Float16)val;
      }
    }
  }
}

// ---------------------------------------------------------------- flash attn
// One wave per 16-row query tile; streams 32 keys/iter (2 WMMA score tiles),
// online softmax, P routed through LDS into A-fragment layout, 4 P*V WMMAs.
__global__ __launch_bounds__(256)
void flash_attn(const _Float16* __restrict__ Q,   // [B*H, S, 64]
                const _Float16* __restrict__ Kc,  // [B*H, S, 64]
                const _Float16* __restrict__ Vt,  // [B*H, 64, S]
                _Float16* __restrict__ ctxh) {    // [B*S, 768] f16
  __shared__ __align__(16) _Float16 Pl[8][16][32];  // per-wave P tile

  const int tid  = threadIdx.x;
  const int wv   = tid >> 5;
  const int lane = tid & 31;
  const int lr   = lane & 15;
  const int hi   = lane >> 4;

  const int bh = blockIdx.y;                     // 0..23
  const int q0 = (blockIdx.x * 8 + wv) * 16;     // query tile base
  const size_t qkbase = (size_t)bh * SEQ * D_K;
  const size_t vbase  = (size_t)bh * D_K * SEQ;

  // Q A-fragments: dk 0..31 and 32..63
  AFrag qa0, qa1;
  {
    const _Float16* qrow = Q + qkbase + (size_t)(q0 + lr) * D_K;
    qa0.q[0] = *(const uint4*)(qrow + hi * 8);
    qa0.q[1] = *(const uint4*)(qrow + 16 + hi * 8);
    qa1.q[0] = *(const uint4*)(qrow + 32 + hi * 8);
    qa1.q[1] = *(const uint4*)(qrow + 48 + hi * 8);
  }

  CFrag ctx[4];
#pragma unroll
  for (int dt = 0; dt < 4; ++dt)
#pragma unroll
    for (int i = 0; i < 8; ++i) ctx[dt].f[i] = 0.0f;
  float mrow[8], lrow[8];
#pragma unroll
  for (int r = 0; r < 8; ++r) { mrow[r] = -1e30f; lrow[r] = 0.0f; }

  const int qmax = q0 + 15;
  for (int jj = 0; jj <= qmax; jj += 32) {
    // ---- scores: two 16-key tiles, contraction over dk=64 (2 WMMAs each)
    CFrag s[2];
#pragma unroll
    for (int t = 0; t < 2; ++t) {
      int key  = jj + t * 16 + lr;
      int krow = key < SEQ ? key : SEQ - 1;      // clamp; mask kills excess
      const _Float16* kr = Kc + qkbase + (size_t)krow * D_K;
      AFrag kb0, kb1;
      kb0.q[0] = *(const uint4*)(kr + hi * 8);
      kb0.q[1] = *(const uint4*)(kr + 16 + hi * 8);
      kb1.q[0] = *(const uint4*)(kr + 32 + hi * 8);
      kb1.q[1] = *(const uint4*)(kr + 48 + hi * 8);
      CFrag z;
#pragma unroll
      for (int i = 0; i < 8; ++i) z.f[i] = 0.0f;
      s[t].v = __builtin_amdgcn_wmma_f32_16x16x32_f16(
          false, qa0.v, false, kb0.v, (short)0, z.v, false, false);
      s[t].v = __builtin_amdgcn_wmma_f32_16x16x32_f16(
          false, qa1.v, false, kb1.v, (short)0, s[t].v, false, false);
    }

    // ---- causal mask + online softmax (row stats via 16-lane xor shuffles)
#pragma unroll
    for (int r = 0; r < 8; ++r) {
      int query = q0 + r + 8 * hi;
#pragma unroll
      for (int t = 0; t < 2; ++t) {
        int key = jj + t * 16 + lr;
        float v = s[t].f[r] * 0.125f;            // 1/sqrt(64)
        s[t].f[r] = (key > query) ? -1e9f : v;
      }
      float vmax = fmaxf(s[0].f[r], s[1].f[r]);
#pragma unroll
      for (int o = 1; o < 16; o <<= 1)
        vmax = fmaxf(vmax, __shfl_xor(vmax, o, 32));
      float mnew = fmaxf(mrow[r], vmax);
      float corr = __expf(mrow[r] - mnew);
      float rsum = 0.0f;
#pragma unroll
      for (int t = 0; t < 2; ++t) {
        float p = __expf(s[t].f[r] - mnew);
        s[t].f[r] = p;
        rsum += p;
      }
#pragma unroll
      for (int o = 1; o < 16; o <<= 1)
        rsum += __shfl_xor(rsum, o, 32);
      lrow[r] = lrow[r] * corr + rsum;
      mrow[r] = mnew;
#pragma unroll
      for (int dt = 0; dt < 4; ++dt) ctx[dt].f[r] *= corr;

      int prow = r + 8 * hi;                     // C layout -> LDS row
      Pl[wv][prow][lr]      = (_Float16)s[0].f[r];
      Pl[wv][prow][16 + lr] = (_Float16)s[1].f[r];
    }
    // same-wave cross-lane LDS store->load: wait DS ops (CDNA5 split counter)
    asm volatile("s_wait_dscnt 0" ::: "memory");

    // ---- P (A-frag from LDS) x V (B-frags from transposed V, contiguous)
    AFrag pa;
    pa.q[0] = *(const uint4*)&Pl[wv][lr][hi * 8];
    pa.q[1] = *(const uint4*)&Pl[wv][lr][16 + hi * 8];
#pragma unroll
    for (int dt = 0; dt < 4; ++dt) {
      const _Float16* vr = Vt + vbase + (size_t)(dt * 16 + lr) * SEQ;
      int c0 = jj + hi * 8;      if (c0 > SEQ - 8) c0 = SEQ - 8;
      int c1 = jj + 16 + hi * 8; if (c1 > SEQ - 8) c1 = SEQ - 8;
      AFrag vb;
      vb.q[0] = *(const uint4*)(vr + c0);
      vb.q[1] = *(const uint4*)(vr + c1);
      ctx[dt].v = __builtin_amdgcn_wmma_f32_16x16x32_f16(
          false, pa.v, false, vb.v, (short)0, ctx[dt].v, false, false);
    }
  }

  // ---- normalize and emit ctx as f16 [B,S,H*dk] for the output projection
  const int b = bh / N_HEADS, h = bh % N_HEADS;
#pragma unroll
  for (int r = 0; r < 8; ++r) {
    float inv = 1.0f / lrow[r];
    int srow = q0 + r + 8 * hi;
    size_t base = ((size_t)(b * SEQ + srow)) * D_MODEL + h * D_K;
#pragma unroll
    for (int dt = 0; dt < 4; ++dt)
      ctxh[base + dt * 16 + lr] = (_Float16)(ctx[dt].f[r] * inv);
  }
}

// ----------------------------------------------------------------- launcher
extern "C" void kernel_launch(void* const* d_in, const int* in_sizes, int n_in,
                              void* d_out, int out_size, void* d_ws,
                              size_t ws_size, hipStream_t stream) {
  const float* x  = (const float*)d_in[0];
  // d_in[1] is the causal tril mask; applied analytically in flash_attn.
  const float* wq = (const float*)d_in[2];
  const float* bq = (const float*)d_in[3];
  const float* wk = (const float*)d_in[4];
  const float* bk = (const float*)d_in[5];
  const float* wv = (const float*)d_in[6];
  const float* bv = (const float*)d_in[7];
  const float* wo = (const float*)d_in[8];
  const float* bo = (const float*)d_in[9];

  const size_t XH = (size_t)M_TOT * D_MODEL;    // 3,145,728 halves
  const size_t WH = (size_t)D_MODEL * D_MODEL;  //   589,824 halves
  _Float16* ws  = (_Float16*)d_ws;              // ~34.5 MB total
  _Float16* xh  = ws;  ws += XH;
  _Float16* wqT = ws;  ws += WH;                // weights stored [N,K] f16
  _Float16* wkT = ws;  ws += WH;
  _Float16* wvT = ws;  ws += WH;
  _Float16* woT = ws;  ws += WH;
  _Float16* Qh  = ws;  ws += XH;
  _Float16* Kh  = ws;  ws += XH;
  _Float16* Vth = ws;  ws += XH;
  _Float16* ch  = ws;  ws += XH;

  cvt_f32_f16<<<512, 256, 0, stream>>>(x, xh, (int)XH);
  cvt_transpose_f32_f16<<<256, 256, 0, stream>>>(wq, wqT);
  cvt_transpose_f32_f16<<<256, 256, 0, stream>>>(wk, wkT);
  cvt_transpose_f32_f16<<<256, 256, 0, stream>>>(wv, wvT);
  cvt_transpose_f32_f16<<<256, 256, 0, stream>>>(wo, woT);

  dim3 g(M_TOT / 128, D_MODEL / 64);  // 32 x 12
  gemm_wmma<<<g, 256, 0, stream>>>(xh, wqT, bq, Qh,  0);
  gemm_wmma<<<g, 256, 0, stream>>>(xh, wkT, bk, Kh,  0);
  gemm_wmma<<<g, 256, 0, stream>>>(xh, wvT, bv, Vth, 1);

  flash_attn<<<dim3(SEQ / 128, BATCH * N_HEADS), 256, 0, stream>>>(Qh, Kh, Vth,
                                                                   ch);

  gemm_wmma<<<g, 256, 0, stream>>>(ch, woT, bo, d_out, 2);
}
